// CausalSelfAttention_40123584479786
// MI455X (gfx1250) — matrix operand order repaired and checked by reading, not compiled, not run on hardware
//
#include <hip/hip_runtime.h>
#include <hip/hip_bf16.h>

// B=4, T=2048, D=1024, H=16, HD=64
#define Bb 4
#define Tt 2048
#define Dd 1024
#define Hh 16
#define HD 64

typedef __attribute__((ext_vector_type(16))) __bf16 v16bf;
typedef __attribute__((ext_vector_type(8)))  float  v8f;

union ABf { v16bf v; uint4 q[2]; };

static __device__ __forceinline__ __bf16 f2bf(float f) {
    union { float f; unsigned u; } x; x.f = f;
    unsigned r = x.u + 0x7FFFu + ((x.u >> 16) & 1u);   // round-to-nearest-even
    union { unsigned short s; __bf16 b; } y;
    y.s = (unsigned short)(r >> 16);
    return y.b;
}

static __device__ __forceinline__ v8f wmma_bf16(v16bf a, v16bf b, v8f c) {
    return __builtin_amdgcn_wmma_f32_16x16x32_bf16(false, a, false, b, (short)0, c,
                                                   false, false);
}

// ---- CDNA5 async memory->LDS (ASYNCcnt path, no VGPR staging) ----
// flat shared pointer: LDS byte offset lives in addr[31:0] (ISA 10.2 aperture rules)
static __device__ __forceinline__ unsigned lds_off(const void* p) {
    return (unsigned)(unsigned long long)(uintptr_t)p;
}
static __device__ __forceinline__ void async_load_b128(unsigned lds, const void* gptr) {
    asm volatile("global_load_async_to_lds_b128 %0, %1, off"
                 :: "v"(lds), "v"((unsigned long long)(uintptr_t)gptr)
                 : "memory");
}
static __device__ __forceinline__ void wait_async0() {
    asm volatile("s_wait_asynccnt 0x0" ::: "memory");
}

// ---------------- conversion kernels ----------------
__global__ void cvt_kernel(const float* __restrict__ in, __bf16* __restrict__ out, int n) {
    int i = blockIdx.x * blockDim.x + threadIdx.x;
    if (i < n) out[i] = f2bf(in[i]);
}

// out[n*K + k] = bf16(in[k*N + n])  (transpose K x N -> N x K)
__global__ void cvt_t_kernel(const float* __restrict__ in, __bf16* __restrict__ out,
                             int K, int N) {
    int i = blockIdx.x * blockDim.x + threadIdx.x;
    if (i < K * N) {
        int nn = i / K, kk = i - nn * K;
        out[i] = f2bf(in[(size_t)kk * N + nn]);
    }
}

// ---------------- WMMA GEMM:  C[M,N] = A[M,K] (bf16) @ Bt[N,K]^T (bf16) ----------------
// Block tile 128(M) x 64(N), K-step 32. 256 threads = 8 waves, each wave 32x32.
// Double-buffered LDS, async global->LDS copies overlap with WMMA compute.
// MODE 0: QKV projection epilogue -> scatter Q,K ([B,H,T,64]) and V transposed ([B,H,64,T])
// MODE 1: fp32 store to Of[M x 1024]
template <int MODE>
__global__ __launch_bounds__(256) void gemm_kernel(
    const __bf16* __restrict__ A, const __bf16* __restrict__ Bt,
    __bf16* __restrict__ Qb, __bf16* __restrict__ Kb, __bf16* __restrict__ Vt,
    float* __restrict__ Of, int K)
{
    __shared__ __bf16 As[2][128][40];   // +8 pad, rows stay 16B aligned (80B stride)
    __shared__ __bf16 Bs[2][64][32];    // transposed weight tile: Bs[n][k]

    const int tid  = threadIdx.x;
    const int lane = tid & 31, wave = tid >> 5;
    const int half = lane >> 4, l15 = lane & 15;
    const int wm = wave & 3, wn = wave >> 2;           // 4 M-slots x 2 N-slots
    const int mbase = blockIdx.x * 128;
    const int nbase = blockIdx.y * 64;

    // per-thread async copy assignments (fixed for all K-slices)
    const int ar = tid >> 1,  acg = (tid & 1) * 16;    // A: 2 x B128 per thread
    const int bn = tid >> 2,  bkg = (tid & 3) * 8;     // B: 1 x B128 per thread
    const __bf16* aSrcRow = A  + (size_t)(mbase + ar) * K + acg;
    const __bf16* bSrcRow = Bt + (size_t)(nbase + bn) * K + bkg;

    v8f acc[2][2] = {};

    // prologue: stream first K-slice into buffer 0
    async_load_b128(lds_off(&As[0][ar][acg]),     aSrcRow);
    async_load_b128(lds_off(&As[0][ar][acg + 8]), aSrcRow + 8);
    async_load_b128(lds_off(&Bs[0][bn][bkg]),     bSrcRow);

    for (int k0 = 0; k0 < K; k0 += 32) {
        const int cur = (k0 >> 5) & 1;
        wait_async0();
        __syncthreads();
        if (k0 + 32 < K) {                 // stream next slice into the other buffer
            const int nxt = 1 - cur;
            async_load_b128(lds_off(&As[nxt][ar][acg]),     aSrcRow + k0 + 32);
            async_load_b128(lds_off(&As[nxt][ar][acg + 8]), aSrcRow + k0 + 40);
            async_load_b128(lds_off(&Bs[nxt][bn][bkg]),     bSrcRow + k0 + 32);
        }

        ABf a[2], b[2];
        #pragma unroll
        for (int mf = 0; mf < 2; ++mf) {
            int row = wm * 32 + mf * 16 + l15;
            a[mf].q[0] = *reinterpret_cast<const uint4*>(&As[cur][row][8 * half]);
            a[mf].q[1] = *reinterpret_cast<const uint4*>(&As[cur][row][16 + 8 * half]);
        }
        #pragma unroll
        for (int nf = 0; nf < 2; ++nf) {
            int n = wn * 32 + nf * 16 + l15;
            b[nf].q[0] = *reinterpret_cast<const uint4*>(&Bs[cur][n][8 * half]);
            b[nf].q[1] = *reinterpret_cast<const uint4*>(&Bs[cur][n][16 + 8 * half]);
        }
        #pragma unroll
        for (int mf = 0; mf < 2; ++mf)
            #pragma unroll
            for (int nf = 0; nf < 2; ++nf)
                acc[mf][nf] = wmma_bf16(a[mf].v, b[nf].v, acc[mf][nf]);
    }

    // epilogue: C layout -> lane holds col (lane&15), rows j + 8*(lane>>4)
    #pragma unroll
    for (int mf = 0; mf < 2; ++mf)
        #pragma unroll
        for (int nf = 0; nf < 2; ++nf)
            #pragma unroll
            for (int j = 0; j < 8; ++j) {
                int grow = mbase + wm * 32 + mf * 16 + 8 * half + j;
                int gcol = nbase + wn * 32 + nf * 16 + l15;
                float v = acc[mf][nf][j];
                if (MODE == 0) {
                    int s = gcol >> 10, rem = gcol & 1023;
                    int h = rem >> 6,   d   = rem & 63;
                    int bb = grow >> 11, t = grow & 2047;  // T=2048
                    __bf16 bv = f2bf(v);
                    size_t bh = (size_t)(bb * Hh + h);
                    if (s == 0)      Qb[(bh * Tt + t) * HD + d] = bv;
                    else if (s == 1) Kb[(bh * Tt + t) * HD + d] = bv;
                    else             Vt[(bh * HD + d) * Tt + t] = bv;  // V transposed
                } else {
                    Of[(size_t)grow * 1024 + gcol] = v;
                }
            }
}

// ---------------- Flash attention (causal, online softmax) ----------------
// 128 threads = 4 waves; block handles 64 query rows of one (b,h); wave -> 16 rows.
// K/V tiles staged with async global->LDS.
__global__ __launch_bounds__(128) void flash_kernel(
    const __bf16* __restrict__ Qb, const __bf16* __restrict__ Kb,
    const __bf16* __restrict__ Vt, __bf16* __restrict__ attnb)
{
    __shared__ __bf16 Ks[32][64];     // [key][hd]
    __shared__ __bf16 Vs[64][32];     // [hd][key]  (source already transposed)
    __shared__ __bf16 Ps[4][16][32];  // per-wave P tile [qrow][key]

    const int tid  = threadIdx.x;
    const int lane = tid & 31, wave = tid >> 5;
    const int half = lane >> 4, l15 = lane & 15;

    const int qt  = blockIdx.x & 31;       // T/64 = 32 q-tiles
    const int bh  = blockIdx.x >> 5;       // b*16 + h
    const int h   = bh & 15;
    const int bb  = bh >> 4;
    const int qt0 = qt * 64;
    const int qw  = qt0 + wave * 16;       // this wave's first query row

    const __bf16* Qp = Qb + (size_t)bh * Tt * HD;
    const __bf16* Kp = Kb + (size_t)bh * Tt * HD;
    const __bf16* Vp = Vt + (size_t)bh * HD * Tt;

    // per-thread async tile assignments
    const int kkey = tid >> 2, khg = (tid & 3) * 16;   // K tile: 2 x B128
    const int vhd  = tid >> 1, vkg = (tid & 1) * 16;   // V tile: 2 x B128

    // Q fragments (A layout): row = lane&15, k = hd per ISA mapping; loaded once
    ABf qa[2];
    {
        const __bf16* qr = Qp + (size_t)(qw + l15) * HD;
        #pragma unroll
        for (int g = 0; g < 2; ++g) {
            qa[g].q[0] = *reinterpret_cast<const uint4*>(qr + 32 * g + 8 * half);
            qa[g].q[1] = *reinterpret_cast<const uint4*>(qr + 32 * g + 16 + 8 * half);
        }
    }

    v8f o[4] = {};
    float m_[8], l_[8];
    #pragma unroll
    for (int j = 0; j < 8; ++j) { m_[j] = -1.0e30f; l_[j] = 0.0f; }
    const float scale = 0.125f;   // 1/sqrt(64)

    for (int kt = 0; kt < qt0 + 64; kt += 32) {
        __syncthreads();   // previous iteration's tile reads are done
        {
            const __bf16* ks = Kp + (size_t)(kt + kkey) * HD + khg;
            async_load_b128(lds_off(&Ks[kkey][khg]),     ks);
            async_load_b128(lds_off(&Ks[kkey][khg + 8]), ks + 8);
            const __bf16* vs = Vp + (size_t)vhd * Tt + kt + vkg;
            async_load_b128(lds_off(&Vs[vhd][vkg]),      vs);
            async_load_b128(lds_off(&Vs[vhd][vkg + 8]),  vs + 8);
        }
        wait_async0();
        __syncthreads();

        // wave-uniform causal skip (force SGPR branch so EXEC is all-ones at WMMA)
        if (__builtin_amdgcn_readfirstlane((int)(kt < qw + 16))) {
            // S = Q @ K^T : two 16x16 col-tiles, HD=64 -> 2 MMAs each
            v8f s[2] = {};
            #pragma unroll
            for (int sg = 0; sg < 2; ++sg) {
                int key = sg * 16 + l15;
                #pragma unroll
                for (int g = 0; g < 2; ++g) {
                    ABf bf_;
                    bf_.q[0] = *reinterpret_cast<const uint4*>(&Ks[key][32 * g + 8 * half]);
                    bf_.q[1] = *reinterpret_cast<const uint4*>(&Ks[key][32 * g + 16 + 8 * half]);
                    s[sg] = wmma_bf16(qa[g].v, bf_.v, s[sg]);
                }
            }
            // scale + causal mask
            #pragma unroll
            for (int sg = 0; sg < 2; ++sg)
                #pragma unroll
                for (int j = 0; j < 8; ++j) {
                    int qr = qw + 8 * half + j;
                    int kc = kt + sg * 16 + l15;
                    s[sg][j] = (kc <= qr) ? s[sg][j] * scale : -1.0e30f;
                }
            // online softmax: row reductions across the 16-lane group
            float alpha[8];
            #pragma unroll
            for (int j = 0; j < 8; ++j) {
                float mx = fmaxf(s[0][j], s[1][j]);
                mx = fmaxf(mx, __shfl_xor(mx, 1));
                mx = fmaxf(mx, __shfl_xor(mx, 2));
                mx = fmaxf(mx, __shfl_xor(mx, 4));
                mx = fmaxf(mx, __shfl_xor(mx, 8));
                float mnew = fmaxf(m_[j], mx);
                alpha[j] = __expf(m_[j] - mnew);
                float p0 = __expf(s[0][j] - mnew);
                float p1 = __expf(s[1][j] - mnew);
                s[0][j] = p0; s[1][j] = p1;
                float rs = p0 + p1;
                rs += __shfl_xor(rs, 1);
                rs += __shfl_xor(rs, 2);
                rs += __shfl_xor(rs, 4);
                rs += __shfl_xor(rs, 8);
                l_[j] = l_[j] * alpha[j] + rs;
                m_[j] = mnew;
            }
            // P (bf16) -> per-wave LDS, C layout -> memory row-major
            #pragma unroll
            for (int sg = 0; sg < 2; ++sg)
                #pragma unroll
                for (int j = 0; j < 8; ++j)
                    Ps[wave][8 * half + j][sg * 16 + l15] = f2bf(s[sg][j]);
            // rescale accumulator (same-wave LDS ops are in-order; no barrier needed)
            #pragma unroll
            for (int f = 0; f < 4; ++f)
                #pragma unroll
                for (int j = 0; j < 8; ++j)
                    o[f][j] *= alpha[j];
            // O += P @ V
            ABf pa;
            {
                const __bf16* pr = &Ps[wave][l15][0];
                pa.q[0] = *reinterpret_cast<const uint4*>(pr + 8 * half);
                pa.q[1] = *reinterpret_cast<const uint4*>(pr + 16 + 8 * half);
            }
            #pragma unroll
            for (int f = 0; f < 4; ++f) {
                ABf vb;
                int hd = f * 16 + l15;
                vb.q[0] = *reinterpret_cast<const uint4*>(&Vs[hd][8 * half]);
                vb.q[1] = *reinterpret_cast<const uint4*>(&Vs[hd][16 + 8 * half]);
                o[f] = wmma_bf16(pa.v, vb.v, o[f]);
            }
        }
    }

    // normalize + store bf16 [B*T, D] for the output projection
    #pragma unroll
    for (int f = 0; f < 4; ++f)
        #pragma unroll
        for (int j = 0; j < 8; ++j) {
            int row = qw + 8 * half + j;
            int col = h * HD + f * 16 + l15;
            float v = o[f][j] / l_[j];
            attnb[((size_t)(bb * Tt + row)) * Dd + col] = f2bf(v);
        }
}

// ---------------- launcher ----------------
extern "C" void kernel_launch(void* const* d_in, const int* in_sizes, int n_in,
                              void* d_out, int out_size, void* d_ws, size_t ws_size,
                              hipStream_t stream) {
    (void)in_sizes; (void)n_in; (void)out_size; (void)ws_size;
    const float* x    = (const float*)d_in[0];   // [4,2048,1024]
    const float* Wqkv = (const float*)d_in[1];   // [1024,3072]
    const float* Wout = (const float*)d_in[2];   // [1024,1024]
    float*       out  = (float*)d_out;           // [4,2048,1024]

    char* ws = (char*)d_ws;
    size_t o = 0;
    __bf16* xb    = (__bf16*)(ws + o); o += (size_t)Bb * Tt * Dd * 2;        // 16 MB
    __bf16* Wqkvt = (__bf16*)(ws + o); o += (size_t)3 * Dd * Dd * 2;         //  6 MB
    __bf16* Woutt = (__bf16*)(ws + o); o += (size_t)Dd * Dd * 2;             //  2 MB
    __bf16* Qb    = (__bf16*)(ws + o); o += (size_t)Bb * Hh * Tt * HD * 2;   // 16 MB
    __bf16* Kbuf  = (__bf16*)(ws + o); o += (size_t)Bb * Hh * Tt * HD * 2;   // 16 MB
    __bf16* Vtb   = (__bf16*)(ws + o); o += (size_t)Bb * Hh * HD * Tt * 2;   // 16 MB
    __bf16* attnb = (__bf16*)(ws + o); o += (size_t)Bb * Tt * Dd * 2;        // 16 MB

    const int nx = Bb * Tt * Dd;             // 8388608
    cvt_kernel<<<(nx + 255) / 256, 256, 0, stream>>>(x, xb, nx);
    cvt_t_kernel<<<(3 * Dd * Dd + 255) / 256, 256, 0, stream>>>(Wqkv, Wqkvt, Dd, 3 * Dd);
    cvt_t_kernel<<<(Dd * Dd + 255) / 256, 256, 0, stream>>>(Wout, Woutt, Dd, Dd);

    // QKV projection: M=8192, N=3072, K=1024
    gemm_kernel<0><<<dim3(8192 / 128, 3072 / 64), 256, 0, stream>>>(
        xb, Wqkvt, Qb, Kbuf, Vtb, nullptr, Dd);

    // flash attention: one block per (b,h,qtile64)
    flash_kernel<<<Bb * Hh * (Tt / 64), 128, 0, stream>>>(Qb, Kbuf, Vtb, attnb);

    // output projection: M=8192, N=1024, K=1024
    gemm_kernel<1><<<dim3(8192 / 128, 1024 / 64), 256, 0, stream>>>(
        attnb, Woutt, nullptr, nullptr, nullptr, out, Dd);
}
